// LocalGlobalCausalSelfAttention_16037407883515
// MI455X (gfx1250) — matrix operand order repaired and checked by reading, not compiled, not run on hardware
//
#include <hip/hip_runtime.h>

// ---------- types ----------
typedef __attribute__((ext_vector_type(16))) __bf16 v16bf;
typedef __attribute__((ext_vector_type(8)))  __bf16 v8bf;
typedef __attribute__((ext_vector_type(8)))  float  v8f;

__device__ __forceinline__ __bf16 f2bf(float f) {
  unsigned u = __builtin_bit_cast(unsigned, f);
  unsigned r = u + 0x7FFFu + ((u >> 16) & 1u);   // round-to-nearest-even
  unsigned short h = (unsigned short)(r >> 16);
  return __builtin_bit_cast(__bf16, h);
}

__device__ __forceinline__ v16bf cat8(v8bf lo, v8bf hi) {
  return __builtin_shufflevector(lo, hi, 0,1,2,3,4,5,6,7,8,9,10,11,12,13,14,15);
}

// LDS byte offset of a generic pointer to __shared__ (low 32 bits of the address)
__device__ __forceinline__ unsigned lds_off32(const void* p) {
  return (unsigned)(size_t)p;
}

// CDNA5 async global->LDS 16-byte copy (per-lane addresses), ASYNCcnt-tracked.
__device__ __forceinline__ void async_copy16(const void* g, unsigned lds) {
  asm volatile("global_load_async_to_lds_b128 %0, %1, off"
               :: "v"(lds), "v"((unsigned long long)(size_t)g) : "memory");
}
__device__ __forceinline__ void wait_async0() {
  asm volatile("s_wait_asynccnt 0" ::: "memory");
}

// ---------- fp32 -> bf16 convert ----------
__global__ __launch_bounds__(256) void cvt_bf16_kernel(const float* __restrict__ in,
                                                       __bf16* __restrict__ out, int n) {
  int i = blockIdx.x * 256 + threadIdx.x;
  if (i < n) out[i] = f2bf(in[i]);
}

// ---------- fp32 -> bf16 convert + transpose: in KxN (row-major) -> out NxK ----------
__global__ __launch_bounds__(256) void cvtT_bf16_kernel(const float* __restrict__ in,
                                                        __bf16* __restrict__ out,
                                                        int K, int N) {
  int i = blockIdx.x * 256 + threadIdx.x;   // i = n*K + k (coalesced writes)
  if (i < K * N) {
    int n = i / K, k = i - n * K;
    out[i] = f2bf(in[(size_t)k * N + n]);
  }
}

// ---------- bf16 WMMA GEMM: C(MxN) = A(MxK) @ Bt(NxK)^T, fp32 out ----------
// block = 128 threads (4 waves); block tile 64x64; k-step 32.
// Both tiles are row-major 2D copies -> staged with async global->LDS b128.
__global__ __launch_bounds__(128) void gemm_bf16_kernel(
    const __bf16* __restrict__ A, const __bf16* __restrict__ Bt,
    float* __restrict__ C, int M, int N, int K) {
  __shared__ __bf16 As[64 * 40];    // [m][k], padded stride 40 (80B rows, 16B aligned)
  __shared__ __bf16 Bs[64 * 40];    // [n][k], padded stride 40
  const int t    = threadIdx.x;
  const int lane = t & 31;
  const int w    = t >> 5;
  const int n16  = lane & 15;
  const int hi   = lane >> 4;
  const int m0   = blockIdx.y * 64;
  const int n0   = blockIdx.x * 64;

  const unsigned asB = lds_off32(&As[0]);
  const unsigned bsB = lds_off32(&Bs[0]);

  // 64x32 tile = 256 x 16B chunks; thread t copies chunks t and t+128 of each tile.
  const int r0 = t >> 2,         c0 = (t & 3) * 8;          // elems
  const int r1 = (t + 128) >> 2, c1 = ((t + 128) & 3) * 8;

  v8f acc[4] = {};

  for (int kk = 0; kk < K; kk += 32) {
    async_copy16(A  + (size_t)(m0 + r0) * K + kk + c0, asB + (unsigned)(r0 * 80 + c0 * 2));
    async_copy16(A  + (size_t)(m0 + r1) * K + kk + c1, asB + (unsigned)(r1 * 80 + c1 * 2));
    async_copy16(Bt + (size_t)(n0 + r0) * K + kk + c0, bsB + (unsigned)(r0 * 80 + c0 * 2));
    async_copy16(Bt + (size_t)(n0 + r1) * K + kk + c1, bsB + (unsigned)(r1 * 80 + c1 * 2));
    wait_async0();
    __syncthreads();

    const int mr = w * 16 + n16;
    // A layout: K(e) = e + 16*hi -> contiguous 16 elems of As[mr][*]
    v16bf av = cat8(*(const v8bf*)&As[mr * 40 + 16 * hi],
                    *(const v8bf*)&As[mr * 40 + 16 * hi + 8]);
    #pragma unroll
    for (int nt = 0; nt < 4; nt++) {
      const int nc = nt * 16 + n16;
      // B layout: K(e) = e + 16*hi -> contiguous 16 elems of Bs[nc][*]
      v16bf bv = cat8(*(const v8bf*)&Bs[nc * 40 + 16 * hi],
                      *(const v8bf*)&Bs[nc * 40 + 16 * hi + 8]);
      acc[nt] = __builtin_amdgcn_wmma_f32_16x16x32_bf16(
          false, av, false, bv, (short)0, acc[nt], false, false);
    }
    __syncthreads();
  }

  #pragma unroll
  for (int nt = 0; nt < 4; nt++)
    #pragma unroll
    for (int r = 0; r < 8; r++)
      C[(size_t)(m0 + w * 16 + r + 8 * hi) * N + n0 + nt * 16 + n16] = acc[nt][r];
}

// ---------- fused RMS-norm + RoPE + bf16 store, one wave per (token, head) ----------
__global__ __launch_bounds__(32) void norm_rope_kernel(const float* __restrict__ in,
                                                       __bf16* __restrict__ out,
                                                       int rowdim) {
  const int s = blockIdx.x, head = blockIdx.y, t = threadIdx.x;  // t = 0..31
  const float* row = in + (size_t)s * rowdim + head * 64;
  float x1 = row[t], x2 = row[t + 32];
  float ss = x1 * x1 + x2 * x2;
  ss += __shfl_xor(ss, 16, 32);
  ss += __shfl_xor(ss,  8, 32);
  ss += __shfl_xor(ss,  4, 32);
  ss += __shfl_xor(ss,  2, 32);
  ss += __shfl_xor(ss,  1, 32);
  float rn = rsqrtf(ss * (1.0f / 64.0f) + 1.1920929e-07f);
  x1 *= rn; x2 *= rn;
  // inv_freq = 10000^(-t/32); freq = s * inv_freq
  float freq = (float)s * __expf(-(float)t * (9.210340371976184f / 32.0f));
  float sn, cs; __sincosf(freq, &sn, &cs);
  __bf16* orow = out + (size_t)s * rowdim + head * 64;
  orow[t]      = f2bf(x1 * cs + x2 * sn);
  orow[t + 32] = f2bf(-x1 * sn + x2 * cs);
}

// ---------- flash attention: 1 wave per (head, 16-query tile) ----------
// Q: bf16 [2048][1024], K: bf16 [2048][256], Vt: bf16 [256][2048], Y: bf16 [2048][1024]
__global__ __launch_bounds__(32) void attn_kernel(
    const __bf16* __restrict__ Q, const __bf16* __restrict__ Km,
    const __bf16* __restrict__ Vt, __bf16* __restrict__ Y) {
  __shared__ __bf16 Pl[16 * 40];
  const int h    = blockIdx.x;
  const int q0   = blockIdx.y * 16;
  const int lane = threadIdx.x;
  const int n    = lane & 15;
  const int hi   = lane >> 4;
  const int kvh  = h >> 2;                 // GQA: 4 q-heads per kv-head
  const float scale = 0.125f;              // 1/sqrt(64)

  // Q tile as two A operands (hd chunks 0..31, 32..63)
  const __bf16* Qrow = Q + (size_t)(q0 + n) * 1024 + h * 64;
  v16bf aq0 = cat8(*(const v8bf*)(Qrow + 16 * hi),      *(const v8bf*)(Qrow + 16 * hi + 8));
  v16bf aq1 = cat8(*(const v8bf*)(Qrow + 32 + 16 * hi), *(const v8bf*)(Qrow + 40 + 16 * hi));

  v8f o[4] = {};
  float mrow[8], lrow[8];
  #pragma unroll
  for (int r = 0; r < 8; r++) { mrow[r] = -3.0e38f; lrow[r] = 0.0f; }

  const int qlast  = q0 + 15;
  const int wstart = q0 - 256 + 1;         // earliest window start across the tile

  for (int kb = 0; kb <= qlast; kb += 32) {
    if (kb >= 64 && (kb + 31) < wstart) continue;   // chunk fully masked (gap region)

    // K consumed row-major: B(e) = K[key][c*32 + e + 16*hi] -> contiguous per lane
    const __bf16* K0 = Km + (size_t)(kb + n) * 256 + kvh * 64;
    const __bf16* K1 = Km + (size_t)(kb + 16 + n) * 256 + kvh * 64;
    v16bf bk00 = cat8(*(const v8bf*)(K0 + 16 * hi),      *(const v8bf*)(K0 + 16 * hi + 8));
    v16bf bk01 = cat8(*(const v8bf*)(K0 + 32 + 16 * hi), *(const v8bf*)(K0 + 40 + 16 * hi));
    v16bf bk10 = cat8(*(const v8bf*)(K1 + 16 * hi),      *(const v8bf*)(K1 + 16 * hi + 8));
    v16bf bk11 = cat8(*(const v8bf*)(K1 + 32 + 16 * hi), *(const v8bf*)(K1 + 40 + 16 * hi));

    v8f s0 = {}, s1 = {};
    s0 = __builtin_amdgcn_wmma_f32_16x16x32_bf16(false, aq0, false, bk00, (short)0, s0, false, false);
    s0 = __builtin_amdgcn_wmma_f32_16x16x32_bf16(false, aq1, false, bk01, (short)0, s0, false, false);
    s1 = __builtin_amdgcn_wmma_f32_16x16x32_bf16(false, aq0, false, bk10, (short)0, s1, false, false);
    s1 = __builtin_amdgcn_wmma_f32_16x16x32_bf16(false, aq1, false, bk11, (short)0, s1, false, false);

    __syncthreads();                        // protect Pl from previous iteration's readers
    #pragma unroll
    for (int r = 0; r < 8; r++) {
      const int qi  = q0 + r + 8 * hi;
      const int ka  = kb + n;
      const int kbn = kb + 16 + n;
      const bool aa = (ka  <= qi) && ((ka  >= qi - 255) || (ka  < 64));
      const bool ab = (kbn <= qi) && ((kbn >= qi - 255) || (kbn < 64));
      float fa = aa ? s0[r] * scale : -3.0e38f;
      float fb = ab ? s1[r] * scale : -3.0e38f;
      float mx = fmaxf(fa, fb);
      mx = fmaxf(mx, __shfl_xor(mx, 1, 32));
      mx = fmaxf(mx, __shfl_xor(mx, 2, 32));
      mx = fmaxf(mx, __shfl_xor(mx, 4, 32));
      mx = fmaxf(mx, __shfl_xor(mx, 8, 32));
      float newm  = fmaxf(mrow[r], mx);
      float alpha = __expf(mrow[r] - newm);
      float pa = aa ? __expf(fa - newm) : 0.0f;
      float pb = ab ? __expf(fb - newm) : 0.0f;
      float rs = pa + pb;
      rs += __shfl_xor(rs, 1, 32);
      rs += __shfl_xor(rs, 2, 32);
      rs += __shfl_xor(rs, 4, 32);
      rs += __shfl_xor(rs, 8, 32);
      lrow[r] = lrow[r] * alpha + rs;
      mrow[r] = newm;
      o[0][r] *= alpha; o[1][r] *= alpha; o[2][r] *= alpha; o[3][r] *= alpha;
      Pl[(r + 8 * hi) * 40 + n]      = f2bf(pa);
      Pl[(r + 8 * hi) * 40 + n + 16] = f2bf(pb);
    }
    __syncthreads();

    // P in A-layout (transpose via LDS), then P @ V using transposed V (contiguous rows)
    v16bf ap = cat8(*(const v8bf*)&Pl[n * 40 + 16 * hi],
                    *(const v8bf*)&Pl[n * 40 + 16 * hi + 8]);
    #pragma unroll
    for (int nt = 0; nt < 4; nt++) {
      // B(e) = V[key = kb + e + 16*hi][hd] = Vt[hd][kb + e + 16*hi] -> contiguous
      const __bf16* vrow = Vt + (size_t)(kvh * 64 + nt * 16 + n) * 2048 + kb + 16 * hi;
      v16bf bv = cat8(*(const v8bf*)vrow, *(const v8bf*)(vrow + 8));
      o[nt] = __builtin_amdgcn_wmma_f32_16x16x32_bf16(
          false, ap, false, bv, (short)0, o[nt], false, false);
    }
  }

  #pragma unroll
  for (int r = 0; r < 8; r++) {
    float inv = 1.0f / lrow[r];
    __bf16* yp = Y + (size_t)(q0 + r + 8 * hi) * 1024 + h * 64 + n;
    yp[0]  = f2bf(o[0][r] * inv);
    yp[16] = f2bf(o[1][r] * inv);
    yp[32] = f2bf(o[2][r] * inv);
    yp[48] = f2bf(o[3][r] * inv);
  }
}

// ---------- host launcher ----------
extern "C" void kernel_launch(void* const* d_in, const int* in_sizes, int n_in,
                              void* d_out, int out_size, void* d_ws, size_t ws_size,
                              hipStream_t stream) {
  (void)in_sizes; (void)n_in; (void)out_size; (void)ws_size;
  const float* x  = (const float*)d_in[0];
  const float* Wq = (const float*)d_in[1];
  const float* Wk = (const float*)d_in[2];
  const float* Wv = (const float*)d_in[3];
  const float* Wo = (const float*)d_in[4];

  const size_t MB = 1024ull * 1024ull;
  char* ws = (char*)d_ws;
  __bf16* xb   = (__bf16*)(ws + 0);                    // 4 MB  x bf16 [2048][1024]
  __bf16* wqT  = (__bf16*)(ws + 4 * MB);               // 2 MB  Wq^T  [1024][1024]
  __bf16* wkT  = (__bf16*)(ws + 6 * MB);               // 0.5   Wk^T  [256][1024]
  __bf16* wvT  = (__bf16*)(ws + 6 * MB + 512 * 1024);  // 0.5   Wv^T  [256][1024]
  __bf16* woT  = (__bf16*)(ws + 7 * MB);               // 2 MB  Wo^T  [1024][1024]
  float*  Qf   = (float*)(ws + 9 * MB);                // 8 MB  q fp32 [2048][1024]
  float*  Kf   = (float*)(ws + 17 * MB);               // 2 MB  k fp32 [2048][256]
  float*  Vf   = (float*)(ws + 19 * MB);               // 2 MB  v fp32 [2048][256]
  __bf16* Qb   = (__bf16*)(ws + 21 * MB);              // 4 MB  q bf16 (normed+roped)
  __bf16* Kb   = (__bf16*)(ws + 25 * MB);              // 1 MB  k bf16 (normed+roped)
  __bf16* Vtb  = (__bf16*)(ws + 26 * MB);              // 1 MB  V^T bf16 [256][2048]
  __bf16* Yb   = (__bf16*)(ws + 27 * MB);              // 4 MB  attn out bf16

  // 1) converts (+ one-time weight transposes so GEMM tiles are pure row copies)
  cvt_bf16_kernel <<<(2097152 + 255) / 256, 256, 0, stream>>>(x,  xb, 2097152);
  cvtT_bf16_kernel<<<(1048576 + 255) / 256, 256, 0, stream>>>(Wq, wqT, 1024, 1024);
  cvtT_bf16_kernel<<<( 262144 + 255) / 256, 256, 0, stream>>>(Wk, wkT, 1024,  256);
  cvtT_bf16_kernel<<<( 262144 + 255) / 256, 256, 0, stream>>>(Wv, wvT, 1024,  256);
  cvtT_bf16_kernel<<<(1048576 + 255) / 256, 256, 0, stream>>>(Wo, woT, 1024, 1024);

  // 2) projections (WMMA, async-LDS staged)
  gemm_bf16_kernel<<<dim3(1024 / 64, 2048 / 64), 128, 0, stream>>>(xb, wqT, Qf, 2048, 1024, 1024);
  gemm_bf16_kernel<<<dim3( 256 / 64, 2048 / 64), 128, 0, stream>>>(xb, wkT, Kf, 2048,  256, 1024);
  gemm_bf16_kernel<<<dim3( 256 / 64, 2048 / 64), 128, 0, stream>>>(xb, wvT, Vf, 2048,  256, 1024);

  // 3) rms-norm + rope epilogues; V transposes to [channel][seq] for contiguous PV fetches
  norm_rope_kernel<<<dim3(2048, 16), 32, 0, stream>>>(Qf, Qb, 1024);
  norm_rope_kernel<<<dim3(2048,  4), 32, 0, stream>>>(Kf, Kb,  256);
  cvtT_bf16_kernel<<<(524288 + 255) / 256, 256, 0, stream>>>(Vf, Vtb, 2048, 256);

  // 4) flash attention with local+global mask (WMMA)
  attn_kernel<<<dim3(16, 128), 32, 0, stream>>>(Qb, Kb, Vtb, Yb);

  // 5) output projection straight into d_out (WMMA, fp32 out)
  gemm_bf16_kernel<<<dim3(1024 / 64, 2048 / 64), 128, 0, stream>>>(Yb, woT, (float*)d_out, 2048, 1024, 1024);
}
